// DCNv2Align_18562848653394
// MI455X (gfx1250) — matrix operand order repaired and checked by reading, not compiled, not run on hardware
//
#include <hip/hip_runtime.h>

typedef __attribute__((ext_vector_type(16))) _Float16 v16h;
typedef __attribute__((ext_vector_type(8)))  _Float16 v8h;
typedef __attribute__((ext_vector_type(8)))  float    v8f;

#define Bn   8
#define Cn   128
#define Hn   96
#define Wn   96
#define K2   9
#define KK   1152   // Cn * K2, contraction length
#define HW   9216   // Hn * Wn
#define PXT  32     // pixels per workgroup
#define ROWS 136    // padded LDS column-row stride (f16 elems), 16B-aligned rows

// ---------------------------------------------------------------------------
// Kernel 0: repack weight (co, ci, kh, kw) f32 -> f16 with K reordered to
// k' = tap*128 + ci, row-major [co][k'].  295 KB in d_ws, L2-resident.
// ---------------------------------------------------------------------------
__global__ __launch_bounds__(256)
void dcn_repack_w_f16(const float* __restrict__ w, _Float16* __restrict__ wh) {
    int idx = blockIdx.x * 256 + threadIdx.x;
    if (idx >= Cn * KK) return;
    int co  = idx / KK;
    int k   = idx - co * KK;     // k' = tap*128 + ci
    int tap = k >> 7;
    int ci  = k & 127;
    wh[idx] = (_Float16)w[(co * Cn + ci) * K2 + tap];
}

// ---------------------------------------------------------------------------
// Main fused kernel: bilinear sampling -> LDS column tile -> WMMA GEMM.
// Grid: Bn * 96 * 3 workgroups of 256 threads (8 waves).
// Each WG: batch b, row h, 32-pixel strip; produces out[b, 0:128, h, strip].
// Outer loop over the 9 taps: sample full 32px x 128ci tile (one coefficient
// set per tap), then 4 K=32 WMMA chunks per tap.  Double-buffered tile.
// ---------------------------------------------------------------------------
__global__ __launch_bounds__(256)
void dcn_wmma_f16(const float* __restrict__ x, const float* __restrict__ off,
                  const _Float16* __restrict__ wh, float* __restrict__ out) {
    // 9 taps x 32 px bilinear coefficients
    __shared__ int   sIdx[288][4];
    __shared__ float sWt [288][4];
    // double-buffered column tile: [buf][px][ci], padded rows
    __shared__ __align__(16) _Float16 colB[2][PXT][ROWS];

    const int t     = threadIdx.x;
    const int blk   = blockIdx.x;
    const int b     = blk / 288;
    const int rem   = blk - b * 288;
    const int h     = rem / 3;
    const int wbase = (rem - h * 3) * PXT;

    // ---- precompute bilinear corner offsets + masked weights -------------
    if (t < 288) {
        const int tap = t >> 5;          // 0..8
        const int px  = t & 31;          // 0..31
        const int kh  = tap / 3;
        const int kw  = tap - kh * 3;
        const int wx  = wbase + px;

        const float* op = off + ((size_t)(b * 18 + tap * 2)) * HW + h * Wn + wx;
        const float oy = op[0];
        const float ox = op[HW];         // x-offset channel is +1

        const float py  = (float)(h  - 1 + kh) + oy;
        const float pxf = (float)(wx - 1 + kw) + ox;
        const float y0f = floorf(py);
        const float x0f = floorf(pxf);
        const float dy  = py  - y0f;
        const float dx  = pxf - x0f;
        const int y0 = (int)y0f, x0 = (int)x0f;
        const int y1 = y0 + 1,  x1 = x0 + 1;

        const float vy0 = (y0 >= 0 && y0 < Hn) ? 1.f : 0.f;
        const float vy1 = (y1 >= 0 && y1 < Hn) ? 1.f : 0.f;
        const float vx0 = (x0 >= 0 && x0 < Wn) ? 1.f : 0.f;
        const float vx1 = (x1 >= 0 && x1 < Wn) ? 1.f : 0.f;

        const int cy0 = y0 < 0 ? 0 : (y0 > Hn - 1 ? Hn - 1 : y0);
        const int cy1 = y1 < 0 ? 0 : (y1 > Hn - 1 ? Hn - 1 : y1);
        const int cx0 = x0 < 0 ? 0 : (x0 > Wn - 1 ? Wn - 1 : x0);
        const int cx1 = x1 < 0 ? 0 : (x1 > Wn - 1 ? Wn - 1 : x1);

        sIdx[t][0] = cy0 * Wn + cx0;
        sIdx[t][1] = cy0 * Wn + cx1;
        sIdx[t][2] = cy1 * Wn + cx0;
        sIdx[t][3] = cy1 * Wn + cx1;
        sWt [t][0] = (1.f - dy) * (1.f - dx) * vy0 * vx0;
        sWt [t][1] = (1.f - dy) * dx         * vy0 * vx1;
        sWt [t][2] = dy * (1.f - dx)         * vy1 * vx0;
        sWt [t][3] = dy * dx                 * vy1 * vx1;
    }
    __syncthreads();

    // ---- thread roles ----------------------------------------------------
    // WMMA:     wave wv owns co chunk [wv*16, wv*16+16)
    // Sampling: thread owns pixel (t&31), channels [ (t>>5)*16, +16 )
    const int wv   = t >> 5;
    const int lane = t & 31;
    const int half = lane >> 4;       // 0/1 : K-half of fragment
    const int m    = lane & 15;       // M (A row) / N (B,C column)
    const int px   = t & 31;          // sampling pixel
    const int cg   = t >> 5;          // sampling channel group (16 ch)

    v8f acc0 = {};                    // px 0..15
    v8f acc1 = {};                    // px 16..31

    const _Float16* wrow  = wh + (size_t)(wv * 16 + m) * KK + half * 8;
    const float*    xbase = x  + ((size_t)(b * Cn + cg * 16)) * HW;

    for (int tap = 0; tap < K2; ++tap) {
        const int buf = tap & 1;

        // ---- sample 32px x 128ci tile for this tap ----------------------
        // one coefficient set per thread; 16 consecutive channels so the
        // 64 gathers fold into 4 base pointers + immediate offsets (j*HW*4).
        {
            const int e  = (tap << 5) + px;
            const int i0 = sIdx[e][0], i1 = sIdx[e][1];
            const int i2 = sIdx[e][2], i3 = sIdx[e][3];
            const float w0 = sWt[e][0], w1 = sWt[e][1];
            const float w2 = sWt[e][2], w3 = sWt[e][3];
            const float* p0 = xbase + i0;
            const float* p1 = xbase + i1;
            const float* p2 = xbase + i2;
            const float* p3 = xbase + i3;

            v8h lo, hi;
#pragma unroll
            for (int j = 0; j < 8; ++j) {
                const float v = w0 * p0[j * HW] + w1 * p1[j * HW]
                              + w2 * p2[j * HW] + w3 * p3[j * HW];
                lo[j] = (_Float16)v;
            }
#pragma unroll
            for (int j = 8; j < 16; ++j) {
                const float v = w0 * p0[j * HW] + w1 * p1[j * HW]
                              + w2 * p2[j * HW] + w3 * p3[j * HW];
                hi[j - 8] = (_Float16)v;
            }
            _Float16* dst = &colB[buf][px][cg * 16];
            *(v8h*)(dst)     = lo;   // ds_store_b128
            *(v8h*)(dst + 8) = hi;   // ds_store_b128
        }
        __syncthreads();   // sole barrier per tap (double buffer covers WAR)

        // ---- 4 WMMA chunks (K = tap*128 + c2*32 .. +32) -----------------
#pragma unroll
        for (int c2 = 0; c2 < 4; ++c2) {
            // A fragment: 16-bit 16x32 layout
            // lane<16 : K 0..7 & 16..23 ; lane>=16 : K 8..15 & 24..31
            const _Float16* ap = wrow + (tap * 4 + c2) * 32;
            const v8h alo = *(const v8h*)(ap);
            const v8h ahi = *(const v8h*)(ap + 16);
            const v16h A = __builtin_shufflevector(alo, ahi,
                             0,1,2,3,4,5,6,7,8,9,10,11,12,13,14,15);

            // B fragments: 16-bit 32x16 layout
            // lane<16 : K 0..7 & 16..23, N=lane ; lane>=16 : K 8..15 & 24..31, N=lane-16
            const _Float16* bp0 = &colB[buf][m][c2 * 32 + half * 8];
            const v8h b0lo = *(const v8h*)(bp0);
            const v8h b0hi = *(const v8h*)(bp0 + 16);
            const v16h B0 = __builtin_shufflevector(b0lo, b0hi,
                              0,1,2,3,4,5,6,7,8,9,10,11,12,13,14,15);

            const _Float16* bp1 = &colB[buf][16 + m][c2 * 32 + half * 8];
            const v8h b1lo = *(const v8h*)(bp1);
            const v8h b1hi = *(const v8h*)(bp1 + 16);
            const v16h B1 = __builtin_shufflevector(b1lo, b1hi,
                              0,1,2,3,4,5,6,7,8,9,10,11,12,13,14,15);

            acc0 = __builtin_amdgcn_wmma_f32_16x16x32_f16(
                       false, A, false, B0, (short)0, acc0, false, false);
            acc1 = __builtin_amdgcn_wmma_f32_16x16x32_f16(
                       false, A, false, B1, (short)0, acc1, false, false);
        }
    }

    // ---- epilogue: 16x16 f32 C/D layout ---------------------------------
    // VGPR r : lanes 0-15 -> M=r, N=lane ; lanes 16-31 -> M=r+8, N=lane-16
#pragma unroll
    for (int r = 0; r < 8; ++r) {
        const int co = wv * 16 + r + half * 8;
        float* op = out + ((size_t)(b * Cn + co)) * HW + h * Wn + wbase + m;
        op[0]  = acc0[r];
        op[16] = acc1[r];
    }
}

// ---------------------------------------------------------------------------
extern "C" void kernel_launch(void* const* d_in, const int* in_sizes, int n_in,
                              void* d_out, int out_size, void* d_ws, size_t ws_size,
                              hipStream_t stream) {
    const float* x   = (const float*)d_in[0];
    const float* off = (const float*)d_in[1];
    const float* w   = (const float*)d_in[2];
    float*       out = (float*)d_out;
    _Float16*    wh  = (_Float16*)d_ws;   // 128*1152*2 = 294,912 bytes

    dcn_repack_w_f16<<<(Cn * KK + 255) / 256, 256, 0, stream>>>(w, wh);
    dcn_wmma_f16<<<Bn * Hn * (Wn / PXT), 256, 0, stream>>>(x, off, wh, out);
}
